// DecoderSingleEval_42992622633757
// MI455X (gfx1250) — matrix operand order repaired and checked by reading, not compiled, not run on hardware
//
#include <hip/hip_runtime.h>

// ---------------------------------------------------------------------------
// CDNA5 (gfx1250) implementation of the GNN decoder forward pass.
// All GEMMs run through v_wmma_f32_16x16x32_bf16 (bf16 inputs, f32 accum).
// ~1.06 TFLOP of GEMM work vs ~1.15 GB mandatory output bytes => compute
// bound => matrix cores. Wave tile 32x64 (8 WMMAs : 6 LDS frag loads),
// double-buffered LDS staging with packed b64 stores.
// ---------------------------------------------------------------------------

typedef __bf16 bf16;
typedef __attribute__((ext_vector_type(8)))  bf16  v8bf;
typedef __attribute__((ext_vector_type(16))) bf16  v16bf;
typedef __attribute__((ext_vector_type(8)))  float v8f;

// Problem constants (fixed by the reference).
static constexpr int       BGR    = 128;     // graphs
static constexpr int       NXPG   = 1024;    // encoder nodes / graph
static constexpr int       NYPG   = 512;     // decoder nodes / graph
static constexpr long long NX_TOT = 131072;
static constexpr long long NY_TOT = 65536;
static constexpr int       F      = 256;
static constexpr int       EMB    = 256;
static constexpr int       H      = 512;
static constexpr int       VOCAB  = 512;
static constexpr int       NSRT   = 7;
static constexpr long long NEDGE  = NY_TOT * 4;

// ---------------------------------------------------------------------------
// Generic tiled WMMA GEMM: C[z] = epilogue( A[z] @ B[z] )
//   tile 64x128x32, 128 threads = 4 waves (wave32), each wave -> 32x64
//   (2x4 WMMA 16x16 tiles). A is f32 [M,lda] (optionally row-gathered via
//   gidx); B is f32 [K,ldb] (or [N,ldb] when TRANS_B). f32 -> bf16 while
//   staging through double-buffered LDS (packed 4xbf16 b64 stores).
// Epilogue: v = acc; if(accum) v += C; v *= scale; v += bias[n]; v += add; relu.
// All M,N,K used here are multiples of 64/128/32 -> no bounds checks, EXEC
// all-1 inside the WMMA loop as the ISA requires.
// ---------------------------------------------------------------------------
struct GemmP {
  const float* A; int lda; long long bsA;
  const float* B; int ldb; long long bsB;
  float*       C; int ldc; long long bsC;
  const float* add; long long bsAdd;
  const float* bias;
  const int*   gidx; int gstride; int goff;
  int M, N, K;
  float scale;
  int accumulate;
  int relu;
};

__device__ __forceinline__ unsigned pk2(float a, float b) {
  unsigned short ua = __builtin_bit_cast(unsigned short, (bf16)a);
  unsigned short ub = __builtin_bit_cast(unsigned short, (bf16)b);
  return (unsigned)ua | ((unsigned)ub << 16);
}

__device__ __forceinline__ v16bf load_frag16(const bf16* p0, const bf16* p1) {
  v8bf lo = *(const v8bf*)p0;           // 16B-aligned LDS chunk
  v8bf hi = *(const v8bf*)p1;           // 16B-aligned LDS chunk
  return __builtin_shufflevector(lo, hi, 0,1,2,3,4,5,6,7,8,9,10,11,12,13,14,15);
}

template<bool TRANS_B, bool GATHER_A>
__global__ __launch_bounds__(128)
void gemm_wmma_bf16(GemmP p)
{
  // row stride 40 bf16 = 80 B -> every 8-element chunk is 16B aligned,
  // every 4-element chunk 8B aligned (for packed uint2 stores).
  __shared__ __align__(16) bf16 As[2][64][40];    // [buf][m][k]
  __shared__ __align__(16) bf16 Bs[2][128][40];   // [buf][n][k]  (n-major!)

  const int tid  = threadIdx.x;
  const int lane = tid & 31;
  const int wave = tid >> 5;
  const int wm   = (wave >> 1) * 32;     // wave's 32x64 subtile
  const int wn   = (wave & 1)  * 64;
  const long long bm0 = (long long)blockIdx.x * 64;
  const long long bn0 = (long long)blockIdx.y * 128;
  const long long z   = blockIdx.z;

  const float* A  = p.A + z * p.bsA;
  const float* Bm = p.B + z * p.bsB;
  float*       C  = p.C + z * p.bsC;

  // K-invariant A row mapping (gather indices hoisted out of the K loop).
  long long arow[4];
#pragma unroll
  for (int it = 0; it < 4; ++it) {
    int s = tid + 128 * it;              // 512 float4 slots: r = s>>3, c4 = s&7
    long long gm = bm0 + (s >> 3);
    arow[it] = GATHER_A ? (long long)p.gidx[gm * p.gstride + p.goff] : gm;
  }

  v8f acc[2][4] = {};

  const int hl  = lane >> 4;      // half-wave select
  const int l15 = lane & 15;
  const int akb = hl * 8;         // A frag K-base (ISA 16-bit A layout)
  const int bkb = hl * 8;         // B frag K-base (ISA 16-bit B layout)

  const int nk = p.K / 32;

  auto stage = [&](int kt, int buf) {
    const int k0 = kt * 32;
    // ---- A tile (64x32): float4 load -> packed b64 LDS store ----
#pragma unroll
    for (int it = 0; it < 4; ++it) {
      int s  = tid + 128 * it;
      int r  = s >> 3;
      int c4 = (s & 7) * 4;
      const float4 va = *(const float4*)(A + arow[it] * (long long)p.lda + k0 + c4);
      *(uint2*)&As[buf][r][c4] = make_uint2(pk2(va.x, va.y), pk2(va.z, va.w));
    }
    // ---- B tile (128 n x 32 k) into [n][k] ----
    if (TRANS_B) {
      // B stored [N, ldb>=K] row-major: contiguous in k.
#pragma unroll
      for (int it = 0; it < 8; ++it) {
        int s  = tid + 128 * it;           // 1024 float4 slots
        int n  = s >> 3;
        int c4 = (s & 7) * 4;
        const float4 vb = *(const float4*)(Bm + (bn0 + n) * (long long)p.ldb + k0 + c4);
        *(uint2*)&Bs[buf][n][c4] = make_uint2(pk2(vb.x, vb.y), pk2(vb.z, vb.w));
      }
    } else {
      // B stored [K, ldb>=N] row-major: 4 k-strided lane-coalesced loads per
      // thread, one packed contiguous b64 store into the n-major tile.
#pragma unroll
      for (int it = 0; it < 8; ++it) {
        int s  = tid + 128 * it;           // 1024 slots: n = s&127, k4 = (s>>7)*4
        int n  = s & 127;
        int k4 = (s >> 7) * 4;
        const float* bp = Bm + (long long)(k0 + k4) * p.ldb + bn0 + n;
        float b0 = bp[0];
        float b1 = bp[p.ldb];
        float b2 = bp[2 * (long long)p.ldb];
        float b3 = bp[3 * (long long)p.ldb];
        *(uint2*)&Bs[buf][n][k4] = make_uint2(pk2(b0, b1), pk2(b2, b3));
      }
    }
  };

  stage(0, 0);
  __syncthreads();
  int cur = 0;

  for (int kt = 0; kt < nk; ++kt) {
    if (kt + 1 < nk) stage(kt + 1, cur ^ 1);           // overlap with WMMAs
    if (kt + 2 < nk)                                   // global_prefetch_b8
      __builtin_prefetch(A + arow[0] * (long long)p.lda + (kt + 2) * 32, 0, 0);

    // ---- fragments per the ISA 7.12.2 layouts ----
    v16bf afrag[2], bfrag[4];
#pragma unroll
    for (int tm = 0; tm < 2; ++tm) {
      const bf16* ap = &As[cur][wm + tm * 16 + l15][akb];
      afrag[tm] = load_frag16(ap, ap + 16);      // K {akb..akb+7, akb+16..+23}
    }
#pragma unroll
    for (int tn = 0; tn < 4; ++tn) {
      const bf16* bp = &Bs[cur][wn + tn * 16 + l15][bkb];
      bfrag[tn] = load_frag16(bp, bp + 16);      // K {bkb..bkb+7, bkb+16..+23}
    }
#pragma unroll
    for (int tm = 0; tm < 2; ++tm)
#pragma unroll
      for (int tn = 0; tn < 4; ++tn)
        acc[tm][tn] = __builtin_amdgcn_wmma_f32_16x16x32_bf16(
            false, afrag[tm], false, bfrag[tn], (short)0, acc[tm][tn], false, false);

    __syncthreads();
    cur ^= 1;
  }

  // ---- epilogue (C/D layout: VGPR r -> M = r + 8*hl, N = l15) ----
#pragma unroll
  for (int tm = 0; tm < 2; ++tm) {
#pragma unroll
    for (int tn = 0; tn < 4; ++tn) {
#pragma unroll
      for (int r = 0; r < 8; ++r) {
        long long m = bm0 + wm + tm * 16 + r + 8 * hl;
        long long n = bn0 + wn + tn * 16 + l15;
        long long off = m * (long long)p.ldc + n;
        float v = acc[tm][tn][r];
        if (p.accumulate) v += C[off];
        v *= p.scale;
        if (p.bias) v += p.bias[n];
        if (p.add)  v += p.add[z * p.bsAdd + off];
        if (p.relu) v = fmaxf(v, 0.0f);
        C[off] = v;
      }
    }
  }
}

// ---------------------------------------------------------------------------
// Row softmax over rows of length 1024 (NX per graph). One wave32 per row.
// ---------------------------------------------------------------------------
__global__ __launch_bounds__(256)
void softmax1024(const float* __restrict__ S, float* __restrict__ alpha)
{
  const int lane = threadIdx.x & 31;
  const int wv   = threadIdx.x >> 5;
  const long long row = (long long)blockIdx.x * 8 + wv;
  const float* rs = S + row * 1024;
  float v[32];
  float mx = -3.4e38f;
#pragma unroll
  for (int j = 0; j < 32; ++j) { v[j] = rs[lane + 32 * j]; mx = fmaxf(mx, v[j]); }
#pragma unroll
  for (int o = 16; o > 0; o >>= 1) mx = fmaxf(mx, __shfl_xor(mx, o, 32));
  float sum = 0.0f;
#pragma unroll
  for (int j = 0; j < 32; ++j) { v[j] = __expf(v[j] - mx); sum += v[j]; }
#pragma unroll
  for (int o = 16; o > 0; o >>= 1) sum += __shfl_xor(sum, o, 32);
  const float inv = 1.0f / sum;
  float* ra = alpha + row * 1024;
#pragma unroll
  for (int j = 0; j < 32; ++j) ra[lane + 32 * j] = v[j] * inv;
}

// ---------------------------------------------------------------------------
// Embedding gather: y0[i,:] = table[tgt_y[i],:] (float4 per thread).
// ---------------------------------------------------------------------------
__global__ __launch_bounds__(256)
void embed_gather(const int* __restrict__ ids, const float* __restrict__ tab,
                  float* __restrict__ y0)
{
  long long t = (long long)blockIdx.x * 256 + threadIdx.x;
  long long node = t >> 6;            // EMB=256 -> 64 float4 per row
  int c = (int)(t & 63);
  ((float4*)y0)[node * 64 + c] =
      ((const float4*)tab)[(long long)ids[node] * 64 + c];
}

// ---------------------------------------------------------------------------
// Edge relation head: out[e,:] = concat(y3[src[e]], y3[tgt[e]]) @ Wg + bg.
// N=7 is too small for WMMA; one wave per edge, Wg staged in LDS.
// ---------------------------------------------------------------------------
__global__ __launch_bounds__(256)
void edge_rel(const float* __restrict__ y3, const int* __restrict__ src,
              const int* __restrict__ tgt, const float* __restrict__ Wg,
              const float* __restrict__ bg, float* __restrict__ out)
{
  __shared__ float Ws[512 * NSRT];
  for (int i = threadIdx.x; i < 512 * NSRT; i += 256) Ws[i] = Wg[i];
  __syncthreads();

  const int lane = threadIdx.x & 31;
  const int wv   = threadIdx.x >> 5;
  const long long e = (long long)blockIdx.x * 8 + wv;
  const long long s = src[e], t = tgt[e];

  float acc[NSRT] = {};
  for (int k = lane; k < 512; k += 32) {
    float f = (k < 256) ? y3[s * EMB + k] : y3[t * EMB + (k - 256)];
#pragma unroll
    for (int r = 0; r < NSRT; ++r) acc[r] += f * Ws[k * NSRT + r];
  }
#pragma unroll
  for (int r = 0; r < NSRT; ++r) {
#pragma unroll
    for (int o = 16; o > 0; o >>= 1) acc[r] += __shfl_xor(acc[r], o, 32);
  }
  if (lane == 0) {
#pragma unroll
    for (int r = 0; r < NSRT; ++r) out[e * NSRT + r] = acc[r] + bg[r];
  }
}

// ---------------------------------------------------------------------------
// Host orchestration.
// ---------------------------------------------------------------------------
static inline GemmP mkP(const float* A, int lda, long long bsA,
                        const float* B, int ldb, long long bsB,
                        float* C, int ldc, long long bsC,
                        int M, int N, int K, float scale,
                        int accumulate = 0, int relu = 0,
                        const float* bias = nullptr,
                        const float* add = nullptr, long long bsAdd = 0,
                        const int* gidx = nullptr, int gstride = 0, int goff = 0)
{
  GemmP p;
  p.A = A; p.lda = lda; p.bsA = bsA;
  p.B = B; p.ldb = ldb; p.bsB = bsB;
  p.C = C; p.ldc = ldc; p.bsC = bsC;
  p.add = add; p.bsAdd = bsAdd; p.bias = bias;
  p.gidx = gidx; p.gstride = gstride; p.goff = goff;
  p.M = M; p.N = N; p.K = K; p.scale = scale;
  p.accumulate = accumulate; p.relu = relu;
  return p;
}

static inline void launch_gemm(hipStream_t st, bool transB, bool gatherA,
                               int batches, const GemmP& p)
{
  dim3 grid(p.M / 64, p.N / 128, batches);
  dim3 blk(128);
  if (gatherA)      gemm_wmma_bf16<false, true ><<<grid, blk, 0, st>>>(p);
  else if (transB)  gemm_wmma_bf16<true,  false><<<grid, blk, 0, st>>>(p);
  else              gemm_wmma_bf16<false, false><<<grid, blk, 0, st>>>(p);
}

extern "C" void kernel_launch(void* const* d_in, const int* in_sizes, int n_in,
                              void* d_out, int out_size, void* d_ws, size_t ws_size,
                              hipStream_t stream)
{
  (void)in_sizes; (void)n_in; (void)out_size; (void)ws_size;

  const float* x       = (const float*)d_in[0];
  const int*   tgt_y   = (const int*)d_in[2];
  const int*   eidx    = (const int*)d_in[3];
  const float* table   = (const float*)d_in[6];
  const float* W_et1   = (const float*)d_in[7];
  const float* Wq1 = (const float*)d_in[8];
  const float* Wk1 = (const float*)d_in[9];
  const float* Wv1 = (const float*)d_in[10];
  const float* W_et2   = (const float*)d_in[11];
  const float* Wq2 = (const float*)d_in[12];
  const float* Wk2 = (const float*)d_in[13];
  const float* Wv2 = (const float*)d_in[14];
  const float* W_et3   = (const float*)d_in[15];
  const float* Wq3 = (const float*)d_in[16];
  const float* Wk3 = (const float*)d_in[17];
  const float* Wv3 = (const float*)d_in[18];
  const float* lin_z_w = (const float*)d_in[19];
  const float* lin_z_b = (const float*)d_in[20];
  const float* lin_g_w = (const float*)d_in[21];
  const float* lin_g_b = (const float*)d_in[22];

  const int* src  = eidx;
  const int* tgtv = eidx + NEDGE;

  // d_out layout (floats): y3, y_score, y_edge_rel, embeds, a1, a2, a3
  float* out       = (float*)d_out;
  float* out_y3    = out;
  float* out_score = out_y3    + NY_TOT * EMB;
  float* out_edge  = out_score + NY_TOT * VOCAB;
  float* out_emb   = out_edge  + NEDGE * NSRT;
  float* out_a1    = out_emb   + NY_TOT * VOCAB;
  float* out_a2    = out_a1 + (long long)BGR * NYPG * NXPG;
  float* out_a3    = out_a2 + (long long)BGR * NYPG * NXPG;

  // Workspace layout (floats); total ~352M floats (~1.41 GB)
  float* w   = (float*)d_ws;
  float* y0  = w; w += NY_TOT * EMB;
  float* y1  = w; w += NY_TOT * H;
  float* y2  = w; w += NY_TOT * H;
  float* agg = w; w += NY_TOT * H;
  float* q   = w; w += NY_TOT * H;
  float* kbf = w; w += NX_TOT * H;
  float* vbf = w; w += NX_TOT * H;
  float* S   = w; w += (long long)BGR * NYPG * NXPG;

  // ---- y0 = embed_table[tgt_y] ----
  embed_gather<<<(NY_TOT * 64) / 256, 256, 0, stream>>>(tgt_y, table, y0);

  // ---- embeds = y0 @ lin_z_w + b ----
  launch_gemm(stream, false, false, 1,
      mkP(y0, EMB, 0, lin_z_w, VOCAB, 0, out_emb, VOCAB, 0,
          (int)NY_TOT, VOCAB, EMB, 1.0f, 0, 0, lin_z_b));

  // ---- one decoder layer ----
  auto run_layer = [&](const float* y_in, int din, int dout,
                       const float* Wet, const float* Wq,
                       const float* Wk, const float* Wv,
                       float* alpha_out, float* y_out)
  {
    // agg = 0.25 * sum_t  Y[src_t] @ Wet[t]      (deg == 4 by construction)
    for (int t = 0; t < 4; ++t) {
      launch_gemm(stream, false, true, 1,
          mkP(y_in, din, 0, Wet + (long long)t * din * dout, dout, 0,
              agg, dout, 0, (int)NY_TOT, dout, din,
              (t == 3) ? 0.25f : 1.0f, (t > 0) ? 1 : 0,
              0, nullptr, nullptr, 0, src, 4, t));
    }
    // q = agg @ Wq ; k = x @ Wk ; v = x @ Wv
    launch_gemm(stream, false, false, 1,
        mkP(agg, dout, 0, Wq, dout, 0, q, dout, 0, (int)NY_TOT, dout, dout, 1.0f));
    launch_gemm(stream, false, false, 1,
        mkP(x, F, 0, Wk, dout, 0, kbf, dout, 0, (int)NX_TOT, dout, F, 1.0f));
    launch_gemm(stream, false, false, 1,
        mkP(x, F, 0, Wv, dout, 0, vbf, dout, 0, (int)NX_TOT, dout, F, 1.0f));
    // S[b] = (q_b @ k_b^T) / sqrt(d)   (batched, B transposed)
    const float sc = 1.0f / sqrtf((float)dout);
    launch_gemm(stream, true, false, BGR,
        mkP(q, dout, (long long)NYPG * dout,
            kbf, dout, (long long)NXPG * dout,
            S, NXPG, (long long)NYPG * NXPG,
            NYPG, NXPG, dout, sc));
    // alpha = softmax(S) -> d_out
    softmax1024<<<(BGR * NYPG) / 8, 256, 0, stream>>>(S, alpha_out);
    // y_out[b] = relu(agg_b + alpha_b @ v_b)   (batched, fused residual+relu)
    launch_gemm(stream, false, false, BGR,
        mkP(alpha_out, NXPG, (long long)NYPG * NXPG,
            vbf, dout, (long long)NXPG * dout,
            y_out, dout, (long long)NYPG * dout,
            NYPG, dout, NXPG, 1.0f, 0, 1, nullptr,
            agg, (long long)NYPG * dout));
  };

  run_layer(y0, F,   H,   W_et1, Wq1, Wk1, Wv1, out_a1, y1);
  run_layer(y1, H,   H,   W_et2, Wq2, Wk2, Wv2, out_a2, y2);
  run_layer(y2, H,   EMB, W_et3, Wq3, Wk3, Wv3, out_a3, out_y3);

  // ---- y_score = y3 @ lin_z_w + b ----
  launch_gemm(stream, false, false, 1,
      mkP(out_y3, EMB, 0, lin_z_w, VOCAB, 0, out_score, VOCAB, 0,
          (int)NY_TOT, VOCAB, EMB, 1.0f, 0, 0, lin_z_b));

  // ---- y_edge_rel_score = concat(y3[src], y3[tgt]) @ lin_g_w + b ----
  edge_rel<<<NEDGE / 8, 256, 0, stream>>>(out_y3, src, tgtv, lin_g_w, lin_g_b, out_edge);
}
